// EDSABlock_37718402794029
// MI455X (gfx1250) — matrix operand order repaired and checked by reading, not compiled, not run on hardware
//
#include <hip/hip_runtime.h>
#include <hip/hip_bf16.h>

typedef __attribute__((ext_vector_type(16))) __bf16 v16bf;
typedef __attribute__((ext_vector_type(8)))  float  v8f;

union Ubf { v16bf v; unsigned int u[8]; unsigned short s[16]; };

__device__ __forceinline__ unsigned short f2bf(float f) {
    unsigned int u = __float_as_uint(f);
    u += 0x7FFFu + ((u >> 16) & 1u);          // round-to-nearest-even
    return (unsigned short)(u >> 16);
}
__device__ __forceinline__ float bf2f(unsigned short s) {
    return __uint_as_float(((unsigned int)s) << 16);
}
__device__ __forceinline__ unsigned int pack2bf(float a, float b) {
    return (unsigned int)f2bf(a) | ((unsigned int)f2bf(b) << 16);
}

__device__ __forceinline__ v8f wmma_bf16(v16bf a, v16bf b, v8f c) {
    return __builtin_amdgcn_wmma_f32_16x16x32_bf16(
        false, a, false, b, (short)0, c, false, false);
}

// A fragment (16x32 bf16) from row-major source viewed as dwords.
// strideW = dwords per row. Row clamped to clampRow (which must be a zero row
// when padding is needed). lane l: row m = mt*16+(l&15); K-half by (l>>4).
// dword index: k/2 = v + (l>>4)*4 for v<4 ; 8+(v-4)+(l>>4)*4 for v>=4 ; + ks*16.
__device__ __forceinline__ v16bf loadA_u(const unsigned int* p, int strideW,
                                         int mt, int ks, int l, int clampRow) {
    Ubf u;
    int m = mt * 16 + (l & 15);
    m = (m < clampRow) ? m : clampRow;
    int base = m * strideW + ks * 16 + (l >> 4) * 4;
#pragma unroll
    for (int v = 0; v < 4; ++v) u.u[v] = p[base + v];
#pragma unroll
    for (int v = 4; v < 8; ++v) u.u[v] = p[base + 8 + (v - 4)];
    return u.v;
}

// B fragment (32x16 bf16): B[k][n] = src[(obase+n)*strideW*2 + k] (row-major [N x K]).
// dword index: o*strideW + ks*16 + v + (l>>4)*4 for v<4 ; 8+(v-4)+(l>>4)*4 for v>=4.
__device__ __forceinline__ v16bf loadB_u(const unsigned int* p, int strideW,
                                         int obase, int ks, int l) {
    Ubf u;
    int o = obase + (l & 15);
    int base = o * strideW + ks * 16 + (l >> 4) * 4;
#pragma unroll
    for (int v = 0; v < 4; ++v) u.u[v] = p[base + v];
#pragma unroll
    for (int v = 4; v < 8; ++v) u.u[v] = p[base + 8 + (v - 4)];
    return u.v;
}

#define NWIN 2048
#define NTOK 49
#define DIM  384
#define NH   12
#define HD   32

// ---------------- kernel 1: f32 -> bf16 weight conversion (packed) ----------------
__global__ void conv_w_kernel(const float* __restrict__ qkv_w,
                              const float* __restrict__ proj_w,
                              unsigned int* __restrict__ wq,
                              unsigned int* __restrict__ wp) {
    int g = blockIdx.x * blockDim.x + threadIdx.x;
    int stride = gridDim.x * blockDim.x;
    const int nq = 3 * DIM * DIM / 2;
    const int np = DIM * DIM / 2;
    for (int i = g; i < nq; i += stride)
        wq[i] = pack2bf(qkv_w[2 * i], qkv_w[2 * i + 1]);
    for (int i = g; i < np; i += stride)
        wp[i] = pack2bf(proj_w[2 * i], proj_w[2 * i + 1]);
}

// ---------------- kernel 2: CPB MLP -> bias table [169][12] ----------------
__device__ __forceinline__ float tblval(int i) {
    float r = ((float)(i - 6)) / 6.0f * 8.0f;
    float a = fabsf(r);
    float s = (r > 0.f) ? 1.f : ((r < 0.f) ? -1.f : 0.f);
    return s * log2f(a + 1.0f) / log2f(8.0f);
}

__global__ void bias_mlp_kernel(const float* __restrict__ w1,
                                const float* __restrict__ b1,
                                const float* __restrict__ w2,
                                float* __restrict__ tab) {
    int t = blockIdx.x;            // 0..168
    int ti = t / 13, tj = t % 13;
    float v0 = tblval(ti), v1 = tblval(tj);
    __shared__ float hid[512];
    __shared__ float red[256];
    int tid = threadIdx.x;
    for (int r = tid; r < 512; r += 256)
        hid[r] = fmaxf(v0 * w1[r * 2 + 0] + v1 * w1[r * 2 + 1] + b1[r], 0.0f);
    __syncthreads();
    for (int h = 0; h < NH; ++h) {
        float p = 0.f;
        for (int r = tid; r < 512; r += 256) p += hid[r] * w2[h * 512 + r];
        red[tid] = p; __syncthreads();
        for (int s = 128; s > 0; s >>= 1) {
            if (tid < s) red[tid] += red[tid + s];
            __syncthreads();
        }
        if (tid == 0) tab[t * NH + h] = red[0];
        __syncthreads();
    }
}

// ---------------- kernel 3: expand to 16*sigmoid(bias)[12][49][49] ----------------
__global__ void bias_expand_kernel(const float* __restrict__ tab,
                                   float* __restrict__ bias16) {
    int g = blockIdx.x * blockDim.x + threadIdx.x;
    const int total = NH * NTOK * NTOK;
    if (g >= total) return;
    int h = g / (NTOK * NTOK);
    int rem = g % (NTOK * NTOK);
    int i = rem / NTOK, j = rem % NTOK;
    int rel0 = (i / 7) - (j / 7) + 6;
    int rel1 = (i % 7) - (j % 7) + 6;
    float bv = tab[(rel0 * 13 + rel1) * NH + h];
    bias16[g] = 16.0f / (1.0f + __expf(-bv));
}

// ---------------- kernel 4: fused window attention ----------------
// block = one window, 256 threads = 8 waves.
// LDS layout (63488 B):
//   [0)      xs   bf16 [50][384]  (row 49 zeroed pad)        38400 B
//   [38400)  qh   bf16 [64][32] | kh bf16 [64][32]  -> later P bf16 [64][64]  8192 B
//   [46592)  S    f32  [64][64]  -> later vt bf16 [32][64]   16384 B
//   [62976)  qn f32[64], kn f32[64]                          512 B
__global__ void __launch_bounds__(256)
attn_kernel(const float* __restrict__ x,
            const unsigned int* __restrict__ wq,     // bf16 [1152][384] as dwords
            const float* __restrict__ q_bias,
            const float* __restrict__ v_bias,
            const float* __restrict__ logit_scale,
            const float* __restrict__ bias16,        // [12][49][49]
            float* __restrict__ out) {               // [2048][49][384]
    __shared__ __align__(16) unsigned long long smem_[7936];
    unsigned char* smem = (unsigned char*)smem_;
    unsigned int*   xsu = (unsigned int*)smem;                  // [50][192] dwords
    unsigned short* qh  = (unsigned short*)(smem + 38400);      // [64][32]
    unsigned short* kh  = qh + 64 * HD;                         // [64][32]
    unsigned int*   qhu = (unsigned int*)qh;                    // [64][16]
    unsigned int*   khu = (unsigned int*)kh;
    unsigned int*   Pu  = (unsigned int*)qh;                    // [64][32] dwords (reuse)
    float*          S   = (float*)(smem + 46592);               // [64][64]
    unsigned int*   vtu = (unsigned int*)(smem + 46592);        // [32][32] dwords (reuse)
    float* qn = (float*)(smem + 62976);
    float* kn = qn + 64;

    int b = blockIdx.x;
    int tid = threadIdx.x;
    int w = tid >> 5, l = tid & 31;

    // stage x into LDS as packed bf16; zero pad row 49
    const float* xb = x + (size_t)b * NTOK * DIM;
    for (int i = tid; i < NTOK * DIM / 2; i += 256)
        xsu[i] = pack2bf(xb[2 * i], xb[2 * i + 1]);
    for (int i = tid; i < DIM / 2; i += 256)
        xsu[NTOK * DIM / 2 + i] = 0u;
    __syncthreads();

    for (int h = 0; h < NH; ++h) {
        float scale = __expf(fminf(logit_scale[h], 4.6051702f));

        // ---- q & k tiles: wave w owns tile w of each (mt=w>>1, nt=w&1) ----
        {
            int mt = w >> 1, nt = w & 1;
            v8f accq = {}, acck = {};
            for (int ks = 0; ks < 12; ++ks) {
                v16bf a  = loadA_u(xsu, 192, mt, ks, l, NTOK);
                v16bf bq = loadB_u(wq, 192, h * HD + nt * 16, ks, l);
                v16bf bk = loadB_u(wq, 192, DIM + h * HD + nt * 16, ks, l);
                accq = wmma_bf16(a, bq, accq);
                acck = wmma_bf16(a, bk, acck);
            }
#pragma unroll
            for (int v = 0; v < 8; ++v) {
                int m = mt * 16 + v + 8 * (l >> 4);
                int n = nt * 16 + (l & 15);
                qh[m * HD + n] = f2bf(accq[v] + q_bias[h * HD + n]);
                kh[m * HD + n] = f2bf(acck[v]);
            }
        }
        __syncthreads();

        // ---- row norms ----
        if (tid < 64) {
            float sq = 0.f, sk = 0.f;
            for (int d = 0; d < HD; ++d) {
                float qv = bf2f(qh[tid * HD + d]); sq += qv * qv;
                float kv = bf2f(kh[tid * HD + d]); sk += kv * kv;
            }
            qn[tid] = 1.0f / (sqrtf(sq) + 1e-12f);
            kn[tid] = 1.0f / (sqrtf(sk) + 1e-12f);
        }
        __syncthreads();

        // ---- S = q @ k^T, 16 tiles, wave w owns tiles w and w+8 ----
        for (int j = 0; j < 2; ++j) {
            int st = w + j * 8;
            int mt = st >> 2, nt = st & 3;
            v8f acc = {};
            v16bf a  = loadA_u(qhu, 16, mt, 0, l, 64);
            v16bf bb = loadB_u(khu, 16, nt * 16, 0, l);   // k^T
            acc = wmma_bf16(a, bb, acc);
#pragma unroll
            for (int v = 0; v < 8; ++v) {
                int m = mt * 16 + v + 8 * (l >> 4);
                int n = nt * 16 + (l & 15);
                float val;
                if (m < NTOK && n < NTOK)
                    val = acc[v] * scale * qn[m] * kn[n]
                        + bias16[(h * NTOK + m) * NTOK + n];
                else
                    val = -1e30f;
                S[m * 64 + n] = val;
            }
        }
        __syncthreads();

        // ---- softmax rows -> P (packed bf16, overwrites q/k region) ----
        if (tid < 64) {
            int r = tid;
            if (r < NTOK) {
                float mx = -1e30f;
                for (int c = 0; c < NTOK; ++c) mx = fmaxf(mx, S[r * 64 + c]);
                float sum = 0.f;
                for (int c = 0; c < NTOK; ++c) sum += __expf(S[r * 64 + c] - mx);
                float inv = 1.0f / sum;
                for (int c2 = 0; c2 < 32; ++c2) {
                    int c0 = 2 * c2, c1 = 2 * c2 + 1;
                    float p0 = (c0 < NTOK) ? __expf(S[r * 64 + c0] - mx) * inv : 0.f;
                    float p1 = (c1 < NTOK) ? __expf(S[r * 64 + c1] - mx) * inv : 0.f;
                    Pu[r * 32 + c2] = pack2bf(p0, p1);
                }
            } else {
                for (int c2 = 0; c2 < 32; ++c2) Pu[r * 32 + c2] = 0u;
            }
        }
        __syncthreads();

        // ---- v tiles, stored TRANSPOSED vt[n][m] (overwrites S region) ----
        {
            int mt = w >> 1, nt = w & 1;
            v8f acc = {};
            for (int ks = 0; ks < 12; ++ks) {
                v16bf a  = loadA_u(xsu, 192, mt, ks, l, NTOK);
                v16bf bv = loadB_u(wq, 192, 2 * DIM + h * HD + nt * 16, ks, l);
                acc = wmma_bf16(a, bv, acc);
            }
            int n = nt * 16 + (l & 15);
            int mbase = mt * 16 + 8 * (l >> 4);
            float vb = v_bias[h * HD + n];
#pragma unroll
            for (int v = 0; v < 4; ++v) {
                vtu[(n * 64 + mbase) / 2 + v] =
                    pack2bf(acc[2 * v] + vb, acc[2 * v + 1] + vb);
            }
        }
        __syncthreads();

        // ---- O = P @ v, 8 tiles, stream to global ----
        {
            int mt = w >> 1, nt = w & 1;
            v8f acc = {};
            for (int ks = 0; ks < 2; ++ks) {
                v16bf a  = loadA_u(Pu, 32, mt, ks, l, 64);
                v16bf bb = loadB_u(vtu, 32, nt * 16, ks, l);   // B[k][n] = vt[n][k]
                acc = wmma_bf16(a, bb, acc);
            }
#pragma unroll
            for (int v = 0; v < 8; ++v) {
                int m = mt * 16 + v + 8 * (l >> 4);
                if (m < NTOK) {
                    size_t off = ((size_t)b * NTOK + m) * DIM + h * HD + nt * 16 + (l & 15);
                    out[off] = acc[v];
                }
            }
        }
        __syncthreads();
    }
}

// ---------------- kernel 5: projection, in-place over d_out ----------------
__global__ void __launch_bounds__(256)
proj_kernel(const unsigned int* __restrict__ wp,     // bf16 [384][384] as dwords
            const float* __restrict__ proj_b,
            float* __restrict__ out) {               // [100352][384], in place
    __shared__ __align__(16) unsigned int ysu[64 * DIM / 2];   // 48KB
    int tid = threadIdx.x;
    int w = tid >> 5, l = tid & 31;
    size_t row0 = (size_t)blockIdx.x * 64;
    const float* yb = out + row0 * DIM;

    for (int i = tid; i < 64 * DIM / 2; i += 256)
        ysu[i] = pack2bf(yb[2 * i], yb[2 * i + 1]);
    __syncthreads();

    // 96 tiles (4 mt x 24 nt); wave w does tiles w*12 .. w*12+11
    for (int t = 0; t < 12; ++t) {
        int tile = w * 12 + t;
        int mt = tile / 24, nt = tile % 24;
        v8f acc = {};
        for (int ks = 0; ks < 12; ++ks) {
            v16bf a  = loadA_u(ysu, 192, mt, ks, l, 64);
            v16bf bb = loadB_u(wp, 192, nt * 16, ks, l);
            acc = wmma_bf16(a, bb, acc);
        }
#pragma unroll
        for (int v = 0; v < 8; ++v) {
            int m = mt * 16 + v + 8 * (l >> 4);
            int n = nt * 16 + (l & 15);
            out[(row0 + m) * DIM + n] = acc[v] + proj_b[n];
        }
    }
}

extern "C" void kernel_launch(void* const* d_in, const int* in_sizes, int n_in,
                              void* d_out, int out_size, void* d_ws, size_t ws_size,
                              hipStream_t stream) {
    const float* x           = (const float*)d_in[0];
    const float* qkv_w       = (const float*)d_in[1];
    const float* q_bias      = (const float*)d_in[2];
    const float* v_bias      = (const float*)d_in[3];
    const float* logit_scale = (const float*)d_in[4];
    const float* cpb_w1      = (const float*)d_in[5];
    const float* cpb_b1      = (const float*)d_in[6];
    const float* cpb_w2      = (const float*)d_in[7];
    const float* proj_w      = (const float*)d_in[8];
    const float* proj_b      = (const float*)d_in[9];
    float* out = (float*)d_out;

    char* ws = (char*)d_ws;
    float*        bias16 = (float*)ws;                       // 12*49*49 f32 = 115248 B
    unsigned int* wq     = (unsigned int*)(ws + 115456);     // 1152*384 bf16 = 884736 B
    unsigned int* wp     = (unsigned int*)(ws + 1000192);    // 384*384 bf16  = 294912 B
    float*        tab    = (float*)(ws + 1295104);           // 169*12 f32

    conv_w_kernel<<<512, 256, 0, stream>>>(qkv_w, proj_w, wq, wp);
    bias_mlp_kernel<<<169, 256, 0, stream>>>(cpb_w1, cpb_b1, cpb_w2, tab);
    bias_expand_kernel<<<(NH * NTOK * NTOK + 255) / 256, 256, 0, stream>>>(tab, bias16);
    attn_kernel<<<NWIN, 256, 0, stream>>>(x, wq, q_bias, v_bias, logit_scale, bias16, out);
    proj_kernel<<<(NWIN * NTOK) / 64, 256, 0, stream>>>(wp, proj_b, out);
}
